// GAPnet_54657753809341
// MI455X (gfx1250) — matrix operand order repaired and checked by reading, not compiled, not run on hardware
//
#include <hip/hip_runtime.h>
#include <hip/hip_bf16.h>

typedef __attribute__((ext_vector_type(16))) _Float16 v16h;
typedef __attribute__((ext_vector_type(8)))  float    v8f;

constexpr int N = 8192;

// ---------------------------------------------------------------- helpers
__device__ inline void top10_insert(float (&bd)[10], int (&bj)[10], float d, int j) {
  if (d < bd[9]) {
    bd[9] = d; bj[9] = j;
#pragma unroll
    for (int s = 9; s > 0; --s) {
      if (bd[s] < bd[s - 1]) {
        float td = bd[s]; bd[s] = bd[s - 1]; bd[s - 1] = td;
        int   tj = bj[s]; bj[s] = bj[s - 1]; bj[s - 1] = tj;
      }
    }
  }
}

__device__ inline float block_reduce_max256(float v, float* red) {
  int t = threadIdx.x; red[t] = v; __syncthreads();
  for (int o = 128; o > 0; o >>= 1) {
    if (t < o) red[t] = fmaxf(red[t], red[t + o]);
    __syncthreads();
  }
  float r = red[0]; __syncthreads(); return r;
}
__device__ inline float block_reduce_sum256(float v, float* red) {
  int t = threadIdx.x; red[t] = v; __syncthreads();
  for (int o = 128; o > 0; o >>= 1) {
    if (t < o) red[t] = red[t] + red[t + o];
    __syncthreads();
  }
  float r = red[0]; __syncthreads(); return r;
}

// ---------------------------------------------------------------- operand packing
// Xt (N x KP) f16, zero-padded K rows: Xt[i*KP + k] = (f16)X[k*N + i]
// Consecutive threads -> consecutive k for one i -> coalesced 2B writes per warp
// within a point's row; reads are strided but single-use and L2-resident.
__global__ __launch_bounds__(256)
void cvt_xT(const float* __restrict__ X, _Float16* __restrict__ Xt,
            int Cin, int KP) {
  size_t e = (size_t)blockIdx.x * 256 + threadIdx.x;   // e = i*KP + k
  int i = (int)(e / KP);
  int k = (int)(e % KP);
  Xt[e] = (k < Cin) ? (_Float16)X[(size_t)k * N + i] : (_Float16)0.0f;
}

// PW packed in A-fragment order: [mtile][kstep][lane][e(16)]
//   lo = lane&15, hi = lane>>4, K = 32*ks + 16*hi + e
//   PW[...] = W[(mt*16+lo)*Cin + K]  (0 beyond Cin)
__global__ __launch_bounds__(256)
void pack_W(const float* __restrict__ W, _Float16* __restrict__ PW,
            int Cin, int KP) {
  size_t idx = (size_t)blockIdx.x * 256 + threadIdx.x;  // O*KP total
  int e    = (int)(idx & 15);
  int lane = (int)((idx >> 4) & 31);
  size_t rest = idx >> 9;
  int nk = KP / 32;
  int ks = (int)(rest % nk);
  int mt = (int)(rest / nk);
  int lo = lane & 15, hi = lane >> 4;
  int kk = 32 * ks + 16 * hi + e;
  PW[idx] = (kk < Cin) ? (_Float16)W[(size_t)(mt * 16 + lo) * Cin + kk]
                       : (_Float16)0.0f;
}

// ---------------------------------------------------------------- WMMA GEMM
// Out(O x N) = W(O x Cin) @ X(Cin x N) + bias, f16 inputs / f32 accumulate.
// Wave computes a 16x16 C tile (D layout: M = v + 8*hi, N = lane&15);
// block = 8 waves -> 32(O) x 64(N). Both operands are pre-packed so each
// lane's fragment per k-step is one contiguous 32B load.
template <int KP>
__global__ __launch_bounds__(256)
void gemm_wmma(const _Float16* __restrict__ PW, const float* __restrict__ Bias,
               const _Float16* __restrict__ Xt, float* __restrict__ Out) {
  constexpr int NK = KP / 32;
  const int lane = threadIdx.x & 31;
  const int wave = threadIdx.x >> 5;
  const int hi   = lane >> 4;
  const int lo   = lane & 15;
  const int m0   = blockIdx.y * 32 + (wave >> 2) * 16;
  const int n0   = blockIdx.x * 64 + (wave & 3) * 16;
  const int mt   = m0 >> 4;

  const _Float16* aw = PW + ((size_t)mt * NK * 32 + lane) * 16;
  const _Float16* bx = Xt + (size_t)(n0 + lo) * KP + 16 * hi;

  v8f acc = {};
#pragma unroll
  for (int ks = 0; ks < NK; ++ks) {
    v16h a = *(const v16h*)(aw + (size_t)ks * 32 * 16);
    v16h b = *(const v16h*)(bx + 32 * ks);
    acc = __builtin_amdgcn_wmma_f32_16x16x32_f16(false, a, false, b, (short)0, acc,
                                                 false, false);
  }
#pragma unroll
  for (int v = 0; v < 8; ++v) {
    int m = m0 + v + 8 * hi;
    Out[(size_t)m * N + n0 + lo] = acc[v] + Bias[m];
  }
}

// ---------------------------------------------------------------- KNN (3D)
__global__ __launch_bounds__(256)
void knn1_kernel(const float* __restrict__ x, float* __restrict__ xknn1) {
  __shared__ float sp[2048 * 4];
  int row = blockIdx.x * 256 + threadIdx.x;
  float px = x[row], py = x[N + row], pz = x[2 * N + row];
  float q = px * px + py * py + pz * pz;
  float bd[10]; int bj[10];
#pragma unroll
  for (int s = 0; s < 10; ++s) { bd[s] = 1e30f; bj[s] = 0; }
  for (int c0 = 0; c0 < N; c0 += 2048) {
    __syncthreads();
    for (int t = threadIdx.x; t < 2048; t += 256) {
      float qx = x[c0 + t], qy = x[N + c0 + t], qz = x[2 * N + c0 + t];
      sp[t * 4] = qx; sp[t * 4 + 1] = qy; sp[t * 4 + 2] = qz;
      sp[t * 4 + 3] = qx * qx + qy * qy + qz * qz;
    }
    __syncthreads();
    for (int j = 0; j < 2048; ++j) {
      float d = q + sp[j * 4 + 3]
              - 2.0f * (px * sp[j * 4] + py * sp[j * 4 + 1] + pz * sp[j * 4 + 2]);
      top10_insert(bd, bj, d, c0 + j);
    }
  }
#pragma unroll
  for (int k = 0; k < 10; ++k) {
    int j = bj[k];
    xknn1[(size_t)row * 30 + k * 3 + 0] = x[j];
    xknn1[(size_t)row * 30 + k * 3 + 1] = x[N + j];
    xknn1[(size_t)row * 30 + k * 3 + 2] = x[2 * N + j];
  }
}

// ---------------------------------------------------------------- T-Net (all-linear, collapsed)
__global__ void tnet_compose1(const float* __restrict__ W1, const float* __restrict__ b1,
                              const float* __restrict__ W2, const float* __restrict__ b2,
                              float* __restrict__ W21, float* __restrict__ b21) {
  int o = threadIdx.x;  // 128
  float a0 = 0, a1 = 0, a2 = 0, bb = b2[o];
  for (int c = 0; c < 64; ++c) {
    float w = W2[o * 64 + c];
    a0 += w * W1[c * 3 + 0]; a1 += w * W1[c * 3 + 1]; a2 += w * W1[c * 3 + 2];
    bb += w * b1[c];
  }
  W21[o * 3 + 0] = a0; W21[o * 3 + 1] = a1; W21[o * 3 + 2] = a2; b21[o] = bb;
}

__global__ __launch_bounds__(256)
void tnet_compose2(const float* __restrict__ W3, const float* __restrict__ b3,
                   const float* __restrict__ W21, const float* __restrict__ b21,
                   float* __restrict__ A, float* __restrict__ dv) {
  int o = blockIdx.x * 256 + threadIdx.x;  // 1024
  float a0 = 0, a1 = 0, a2 = 0, bb = b3[o];
  for (int c = 0; c < 128; ++c) {
    float w = W3[o * 128 + c];
    a0 += w * W21[c * 3 + 0]; a1 += w * W21[c * 3 + 1]; a2 += w * W21[c * 3 + 2];
    bb += w * b21[c];
  }
  A[o * 3 + 0] = a0; A[o * 3 + 1] = a1; A[o * 3 + 2] = a2; dv[o] = bb;
}

__global__ __launch_bounds__(256)
void tnet_maxfeat(const float* __restrict__ A, const float* __restrict__ dv,
                  const float* __restrict__ x, float* __restrict__ H) {
  __shared__ float red[256];
  int ch = blockIdx.x;
  float a0 = A[ch * 3], a1 = A[ch * 3 + 1], a2 = A[ch * 3 + 2];
  float m = -1e30f;
  for (int l = threadIdx.x; l < N; l += 256)
    m = fmaxf(m, a0 * x[l] + a1 * x[N + l] + a2 * x[2 * N + l]);
  m = block_reduce_max256(m, red);
  if (threadIdx.x == 0) H[ch] = dv[ch] + m;
}

__global__ __launch_bounds__(256)
void tnet_fc(const float* __restrict__ Wf1, const float* __restrict__ bf1,
             const float* __restrict__ Wf2, const float* __restrict__ bf2,
             const float* __restrict__ Wf3, const float* __restrict__ bf3,
             const float* __restrict__ H, float* __restrict__ t9) {
  __shared__ float h0[1024]; __shared__ float h1[512]; __shared__ float h2[256];
  int t = threadIdx.x;
  for (int o = t; o < 1024; o += 256) h0[o] = H[o];
  __syncthreads();
  for (int o = t; o < 512; o += 256) {
    float a = bf1[o];
    const float* w = Wf1 + (size_t)o * 1024;
    for (int c = 0; c < 1024; ++c) a += w[c] * h0[c];
    h1[o] = a;
  }
  __syncthreads();
  for (int o = t; o < 256; o += 256) {
    float a = bf2[o];
    const float* w = Wf2 + (size_t)o * 512;
    for (int c = 0; c < 512; ++c) a += w[c] * h1[c];
    h2[o] = a;
  }
  __syncthreads();
  if (t < 9) {
    float a = bf3[t];
    const float* w = Wf3 + (size_t)t * 256;
    for (int c = 0; c < 256; ++c) a += w[c] * h2[c];
    t9[t] = a;
  }
}

// xn buffer is (3,N) row-major; reference later views it flat as (N,3).
__global__ __launch_bounds__(256)
void tnet_apply(const float* __restrict__ t9, const float* __restrict__ x,
                float* __restrict__ xn) {
  int l = blockIdx.x * 256 + threadIdx.x;
  int r = blockIdx.y;
  xn[(size_t)r * N + l] =
      t9[r * 3] * x[l] + t9[r * 3 + 1] * x[N + l] + t9[r * 3 + 2] * x[2 * N + l];
}

// ---------------------------------------------------------------- GAP 1 (dim=3, F=16)
__global__ __launch_bounds__(256)
void gap1_a(const float* __restrict__ xknn1, const float* __restrict__ xn,
            const float* __restrict__ WF, const float* __restrict__ bF,
            const float* __restrict__ w1, const float* __restrict__ b1,
            float* __restrict__ feat1, float* __restrict__ sc1) {
  int i = blockIdx.x * 256 + threadIdx.x;
  const float* xk = xknn1 + (size_t)i * 30;  // flat view (3,10)
  float xc0 = xn[3 * i], xc1v = xn[3 * i + 1], xc2 = xn[3 * i + 2];
  float scores[10];
#pragma unroll
  for (int k = 0; k < 10; ++k) scores[k] = 0.0f;
  float selfacc = 0.0f;
  for (int o = 0; o < 16; ++o) {
    float w0 = WF[o * 3], wy = WF[o * 3 + 1], wz = WF[o * 3 + 2], bb = bF[o];
    selfacc += w1[o] * (bb + w0 * xc0 + wy * xc1v + wz * xc2);
#pragma unroll
    for (int k = 0; k < 10; ++k) {
      float fv = bb + w0 * xk[k] + wy * xk[10 + k] + wz * xk[20 + k];
      feat1[(size_t)i * 160 + o * 10 + k] = fv;
      scores[k] += w1[o] * fv;
    }
  }
#pragma unroll
  for (int k = 0; k < 10; ++k)
    sc1[(size_t)i * 10 + k] = scores[k] + selfacc + 2.0f * b1[0];
}

__global__ __launch_bounds__(256)
void softmax_stats(const float* __restrict__ sc, float* __restrict__ m,
                   float* __restrict__ s) {
  __shared__ float red[256];
  int k = blockIdx.x;
  float mv = -1e30f;
  for (int r = threadIdx.x; r < N; r += 256) mv = fmaxf(mv, sc[(size_t)r * 10 + k]);
  float mm = block_reduce_max256(mv, red);
  float sv = 0.0f;
  for (int r = threadIdx.x; r < N; r += 256) sv += expf(sc[(size_t)r * 10 + k] - mm);
  float ss = block_reduce_sum256(sv, red);
  if (threadIdx.x == 0) { m[k] = mm; s[k] = ss; }
}

__global__ __launch_bounds__(256)
void gap1_c(const float* __restrict__ sc1, const float* __restrict__ m,
            const float* __restrict__ s, const float* __restrict__ feat1,
            const float* __restrict__ xn, float* __restrict__ hbuf,
            float* __restrict__ xcbuf) {
  int i = blockIdx.x * 256 + threadIdx.x;
  float xcv[10];
#pragma unroll
  for (int k = 0; k < 10; ++k)
    xcv[k] = expf(sc1[(size_t)i * 10 + k] - m[k]) / s[k];
  hbuf[i] = xn[3 * i]; hbuf[N + i] = xn[3 * i + 1]; hbuf[2 * N + i] = xn[3 * i + 2];
  for (int f = 0; f < 16; ++f) {
    float af = 0.0f, pf = -1e30f;
#pragma unroll
    for (int k = 0; k < 10; ++k) {
      float v = feat1[(size_t)i * 160 + k * 16 + f];  // flat (10,16) view
      af += xcv[k] * v; pf = fmaxf(pf, v);
    }
#pragma unroll
    for (int mh = 0; mh < 4; ++mh) {
      hbuf[(size_t)(3 + f * 4 + mh) * N + i]  = af;   // head_attn -> h concat
      xcbuf[(size_t)(f * 4 + mh) * N + i]     = pf;   // xgraph1_pool
    }
  }
}

// ---------------------------------------------------------------- channel KNN (knn2)
__global__ __launch_bounds__(256)
void chan_sq(const float* __restrict__ xc1, float* __restrict__ sq2) {
  __shared__ float red[256];
  int a = blockIdx.x;
  float sv = 0.0f;
  for (int l = threadIdx.x; l < N; l += 256) {
    float v = xc1[(size_t)a * N + l]; sv += v * v;
  }
  float r = block_reduce_sum256(sv, red);
  if (threadIdx.x == 0) sq2[a] = r;
}

__global__ __launch_bounds__(256)
void gram2(const float* __restrict__ xc1, const float* __restrict__ sq2,
           float* __restrict__ D) {
  __shared__ float red[256];
  int a = blockIdx.x;
  for (int b = 0; b < 128; ++b) {
    float p = 0.0f;
    for (int l = threadIdx.x; l < N; l += 256)
      p += xc1[(size_t)a * N + l] * xc1[(size_t)b * N + l];
    float g = block_reduce_sum256(p, red);
    if (threadIdx.x == 0) D[a * 128 + b] = sq2[a] + sq2[b] - 2.0f * g;
    __syncthreads();
  }
}

__global__ void top10_ch(const float* __restrict__ D, int* __restrict__ idx2) {
  int a = threadIdx.x;  // 128 threads
  float bd[10]; int bj[10];
#pragma unroll
  for (int s = 0; s < 10; ++s) { bd[s] = 1e30f; bj[s] = 0; }
  for (int b = 0; b < 128; ++b) top10_insert(bd, bj, D[a * 128 + b], b);
#pragma unroll
  for (int k = 0; k < 10; ++k) idx2[a * 10 + k] = bj[k];
}

// raw (128,10,8192) flat: raw[a*81920 + k*8192 + l] = xc1[idx2[a][k]*N + l]
__global__ __launch_bounds__(256)
void gather2(const float* __restrict__ xc1, const int* __restrict__ idx2,
             float* __restrict__ raw) {
  size_t e = (size_t)blockIdx.x * 256 + threadIdx.x;
  int a = (int)(e / 81920);
  int k = (int)((e / 8192) % 10);
  int l = (int)(e & 8191);
  raw[e] = xc1[(size_t)idx2[a * 10 + k] * N + l];
}

// ---------------------------------------------------------------- GAP 2 (dim=128, F=128)
__global__ __launch_bounds__(128)
void gap2_a(const float* __restrict__ raw, const float* __restrict__ xc1g,
            const float* __restrict__ W2F, const float* __restrict__ b2F,
            const float* __restrict__ w12, const float* __restrict__ b12,
            float* __restrict__ feat2, float* __restrict__ sc2) {
  __shared__ float xk[1280];    // flat (128,10) view of neighbor block
  __shared__ float xself[128];  // scrambled self vector (bug-compatible)
  __shared__ float cmat[1280];
  __shared__ float sarr[128];
  int i = blockIdx.x, o = threadIdx.x;
  for (int t = o; t < 1280; t += 128) xk[t] = raw[(size_t)i * 1280 + t];
  xself[o] = xc1g[(size_t)i * 128 + o];
  __syncthreads();
  const float* wrow = W2F + (size_t)o * 128;
  float fself = b2F[o];
  float fv[10];
#pragma unroll
  for (int k = 0; k < 10; ++k) fv[k] = b2F[o];
  for (int c = 0; c < 128; ++c) {
    float w = wrow[c];
    fself += w * xself[c];
#pragma unroll
    for (int k = 0; k < 10; ++k) fv[k] += w * xk[c * 10 + k];
  }
  float wo = w12[o];
#pragma unroll
  for (int k = 0; k < 10; ++k) {
    feat2[(size_t)i * 1280 + o * 10 + k] = fv[k];
    cmat[o * 10 + k] = wo * fv[k];
  }
  sarr[o] = wo * fself;
  __syncthreads();
  if (o < 10) {
    float cs = 0.0f, ss = 0.0f;
    for (int oo = 0; oo < 128; ++oo) { cs += cmat[oo * 10 + o]; ss += sarr[oo]; }
    sc2[(size_t)i * 10 + o] = cs + ss + 2.0f * b12[0];
  }
}

__global__ __launch_bounds__(256)
void gap2_c(const float* __restrict__ sc2, const float* __restrict__ m,
            const float* __restrict__ s, const float* __restrict__ feat2,
            float* __restrict__ h2in, float* __restrict__ xcbuf) {
  int i = blockIdx.x * 256 + threadIdx.x;
  float xcv[10];
#pragma unroll
  for (int k = 0; k < 10; ++k)
    xcv[k] = expf(sc2[(size_t)i * 10 + k] - m[k]) / s[k];
  for (int f = 0; f < 128; ++f) {
    float af = 0.0f, pf = -1e30f;
#pragma unroll
    for (int k = 0; k < 10; ++k) {
      float v = feat2[(size_t)i * 1280 + k * 128 + f];  // flat (10,128) view
      af += xcv[k] * v; pf = fmaxf(pf, v);
    }
#pragma unroll
    for (int mh = 0; mh < 4; ++mh) {
      h2in[(size_t)(f * 4 + mh) * N + i]         = af;  // head_attn2 (512,N)
      xcbuf[(size_t)(64 + f * 4 + mh) * N + i]   = pf;  // xgraph2_pool rows 64..575
    }
  }
}

// ---------------------------------------------------------------- pooling + FC head
__global__ __launch_bounds__(256)
void maxpool_k(const float* __restrict__ X, float* __restrict__ out) {
  __shared__ float red[256];
  int o = blockIdx.x;
  float mv = -1e30f;
  for (int l = threadIdx.x; l < N; l += 256) mv = fmaxf(mv, X[(size_t)o * N + l]);
  float r = block_reduce_max256(mv, red);
  if (threadIdx.x == 0) out[o] = r;
}

__global__ __launch_bounds__(256)
void fc_head(const float* __restrict__ W1, const float* __restrict__ b1,
             const float* __restrict__ W2, const float* __restrict__ b2,
             const float* __restrict__ W3, const float* __restrict__ b3,
             const float* __restrict__ W4, const float* __restrict__ b4,
             const float* __restrict__ W5, const float* __restrict__ b5,
             const float* __restrict__ xpool, float* __restrict__ out) {
  __shared__ float bx[2048]; __shared__ float by[1024];
  int t = threadIdx.x;
  for (int o = t; o < 1024; o += 256) by[o] = xpool[o];
  __syncthreads();
  for (int o = t; o < 2048; o += 256) {
    float a = b1[o]; const float* w = W1 + (size_t)o * 1024;
    for (int c = 0; c < 1024; ++c) a += w[c] * by[c];
    bx[o] = a;
  }
  __syncthreads();
  for (int o = t; o < 1024; o += 256) {
    float a = b2[o]; const float* w = W2 + (size_t)o * 2048;
    for (int c = 0; c < 2048; ++c) a += w[c] * bx[c];
    by[o] = a;
  }
  __syncthreads();
  for (int o = t; o < 512; o += 256) {
    float a = b3[o]; const float* w = W3 + (size_t)o * 1024;
    for (int c = 0; c < 1024; ++c) a += w[c] * by[c];
    bx[o] = a;
  }
  __syncthreads();
  for (int o = t; o < 256; o += 256) {
    float a = b4[o]; const float* w = W4 + (size_t)o * 512;
    for (int c = 0; c < 512; ++c) a += w[c] * bx[c];
    by[o] = a;
  }
  __syncthreads();
  if (t < 40) {
    float a = b5[t]; const float* w = W5 + (size_t)t * 256;
    for (int c = 0; c < 256; ++c) a += w[c] * by[c];
    out[t] = a;
  }
}

// ================================================================ host
extern "C" void kernel_launch(void* const* d_in, const int* in_sizes, int n_in,
                              void* d_out, int out_size, void* d_ws, size_t ws_size,
                              hipStream_t stream) {
  enum { TM1 = 0, TM2, TM3, TFC1, TFC2, TFC3, G1F, G11, G2F, G21,
         M1, M2, M3, M4, M5, M6, M7, FC1, FC2, FC3, FC4, FC5 };
  const float* W[22];
  const float* Bv[22];
  const float* xin;
  if (in_sizes[0] == 3 * N) {
    // insertion order: x, then (w, b) per _PARAM_SPECS
    xin = (const float*)d_in[0];
    int idx = 1;
    for (int s = 0; s < 22; ++s) {
      W[s] = (const float*)d_in[idx++];
      Bv[s] = (const float*)d_in[idx++];
    }
  } else {
    // pytree-sorted order: params first (names sorted, 'b' before 'w'), x last
    static const int order[22] = { FC1, FC2, FC3, FC4, FC5, G11, G1F, G21, G2F,
                                   M1, M2, M3, M4, M5, M6, M7,
                                   TFC1, TFC2, TFC3, TM1, TM2, TM3 };
    int idx = 0;
    for (int t = 0; t < 22; ++t) {
      int s = order[t];
      Bv[s] = (const float*)d_in[idx++];
      W[s] = (const float*)d_in[idx++];
    }
    xin = (const float*)d_in[idx];
  }

  // ---- workspace carve
  char* wsb = (char*)d_ws;
  size_t off = 0;
  auto carve = [&](size_t bytes) -> void* {
    void* p = wsb + off;
    off = (off + bytes + 255) & ~(size_t)255;
    return p;
  };
  float* f_xknn1 = (float*)carve((size_t)N * 30 * 4);
  float* f_W21   = (float*)carve(384 * 4);
  float* f_b21   = (float*)carve(128 * 4);
  float* f_A     = (float*)carve(3072 * 4);
  float* f_dv    = (float*)carve(1024 * 4);
  float* f_H     = (float*)carve(1024 * 4);
  float* f_t9    = (float*)carve(16 * 4);
  float* f_xn    = (float*)carve((size_t)3 * N * 4);
  float* f_feat1 = (float*)carve((size_t)N * 160 * 4);
  float* f_sc1   = (float*)carve((size_t)N * 10 * 4);
  float* f_ms    = (float*)carve(64 * 4);         // m1,s1,m2,s2 (10 each)
  float* f_hbuf  = (float*)carve((size_t)67 * N * 4);
  float* f_b64a  = (float*)carve((size_t)64 * N * 4);
  float* f_b64b  = (float*)carve((size_t)64 * N * 4);
  float* f_xc1   = (float*)carve((size_t)128 * N * 4);
  float* f_sq2   = (float*)carve(128 * 4);
  float* f_D     = (float*)carve(128 * 128 * 4);
  int*   i_idx2  = (int*)carve(1280 * 4);
  float* f_raw2  = (float*)carve((size_t)N * 1280 * 4);
  float* f_feat2 = (float*)carve((size_t)N * 1280 * 4);
  float* f_sc2   = (float*)carve((size_t)N * 10 * 4);
  float* f_xcbuf = (float*)carve((size_t)1088 * N * 4);
  float* f_xpool = (float*)carve(1024 * 4);
  _Float16* h_Xt = (_Float16*)carve((size_t)N * 1088 * 2);   // shared f16 activation scratch
  _Float16* h_PW = (_Float16*)carve((size_t)1024 * 1088 * 2);// shared f16 packed-weight scratch
  // aliased reuse of dead regions:
  float* f_h2in  = f_raw2;                 // 512N  (raw2 dead after gap2_a)
  float* f_b128a = f_raw2 + (size_t)512 * N;
  float* f_b128b = f_raw2 + (size_t)640 * N;
  float* f_m7    = f_feat2;                // 1024N (feat2 dead after gap2_c)
  (void)ws_size; (void)n_in; (void)out_size;

  // one packed-operand WMMA GEMM: Out(OxN) = W(OxCin) @ X(CinxN) + b
  auto run_gemm = [&](int KP, int O, int Cin, const float* Wp, const float* bp,
                      const float* Xp, float* Op) {
    cvt_xT<<<(N * KP) / 256, 256, 0, stream>>>(Xp, h_Xt, Cin, KP);
    pack_W<<<(O * KP) / 256, 256, 0, stream>>>(Wp, h_PW, Cin, KP);
    dim3 g(N / 64, O / 32);
    switch (KP) {
      case 64:   gemm_wmma<64><<<g, 256, 0, stream>>>(h_PW, bp, h_Xt, Op); break;
      case 96:   gemm_wmma<96><<<g, 256, 0, stream>>>(h_PW, bp, h_Xt, Op); break;
      case 128:  gemm_wmma<128><<<g, 256, 0, stream>>>(h_PW, bp, h_Xt, Op); break;
      case 512:  gemm_wmma<512><<<g, 256, 0, stream>>>(h_PW, bp, h_Xt, Op); break;
      case 1088: gemm_wmma<1088><<<g, 256, 0, stream>>>(h_PW, bp, h_Xt, Op); break;
    }
  };

  // ---- pipeline
  knn1_kernel<<<N / 256, 256, 0, stream>>>(xin, f_xknn1);
  tnet_compose1<<<1, 128, 0, stream>>>(W[TM1], Bv[TM1], W[TM2], Bv[TM2], f_W21, f_b21);
  tnet_compose2<<<4, 256, 0, stream>>>(W[TM3], Bv[TM3], f_W21, f_b21, f_A, f_dv);
  tnet_maxfeat<<<1024, 256, 0, stream>>>(f_A, f_dv, xin, f_H);
  tnet_fc<<<1, 256, 0, stream>>>(W[TFC1], Bv[TFC1], W[TFC2], Bv[TFC2],
                                 W[TFC3], Bv[TFC3], f_H, f_t9);
  tnet_apply<<<dim3(N / 256, 3), 256, 0, stream>>>(f_t9, xin, f_xn);

  gap1_a<<<N / 256, 256, 0, stream>>>(f_xknn1, f_xn, W[G1F], Bv[G1F],
                                      W[G11], Bv[G11], f_feat1, f_sc1);
  softmax_stats<<<10, 256, 0, stream>>>(f_sc1, f_ms, f_ms + 10);
  gap1_c<<<N / 256, 256, 0, stream>>>(f_sc1, f_ms, f_ms + 10, f_feat1, f_xn,
                                      f_hbuf, f_xcbuf);

  run_gemm(96, 64, 67, W[M1], Bv[M1], f_hbuf, f_b64a);
  run_gemm(64, 64, 64, W[M2], Bv[M2], f_b64a, f_b64b);
  run_gemm(64, 128, 64, W[M3], Bv[M3], f_b64b, f_xc1);

  chan_sq<<<128, 256, 0, stream>>>(f_xc1, f_sq2);
  gram2<<<128, 256, 0, stream>>>(f_xc1, f_sq2, f_D);
  top10_ch<<<1, 128, 0, stream>>>(f_D, i_idx2);
  gather2<<<(128 * 10 * N) / 256, 256, 0, stream>>>(f_xc1, i_idx2, f_raw2);

  gap2_a<<<N, 128, 0, stream>>>(f_raw2, f_xc1, W[G2F], Bv[G2F],
                                W[G21], Bv[G21], f_feat2, f_sc2);
  softmax_stats<<<10, 256, 0, stream>>>(f_sc2, f_ms + 20, f_ms + 30);
  gap2_c<<<N / 256, 256, 0, stream>>>(f_sc2, f_ms + 20, f_ms + 30, f_feat2,
                                      f_h2in, f_xcbuf);

  run_gemm(512, 128, 512, W[M4], Bv[M4], f_h2in, f_b128a);
  run_gemm(128, 128, 128, W[M5], Bv[M5], f_b128a, f_b128b);
  run_gemm(128, 512, 128, W[M6], Bv[M6], f_b128b, f_xcbuf + (size_t)576 * N);
  run_gemm(1088, 1024, 1088, W[M7], Bv[M7], f_xcbuf, f_m7);

  maxpool_k<<<1024, 256, 0, stream>>>(f_m7, f_xpool);
  fc_head<<<1, 256, 0, stream>>>(W[FC1], Bv[FC1], W[FC2], Bv[FC2], W[FC3], Bv[FC3],
                                 W[FC4], Bv[FC4], W[FC5], Bv[FC5],
                                 f_xpool, (float*)d_out);
}
